// encoderLayer_5016521802175
// MI455X (gfx1250) — matrix operand order repaired and checked by reading, not compiled, not run on hardware
//
#include <hip/hip_runtime.h>
#include <cstdint>
#include <cstddef>

// ---------------- problem constants ----------------
constexpr int Bc = 8, Sc = 1024, Dc = 1024, Hc = 16, DKc = 64;
constexpr int Mrows = Bc * Sc;          // 8192
constexpr int QKVW  = 3 * Dc;           // 3072

typedef __attribute__((ext_vector_type(16))) __bf16 v16bf;
typedef __attribute__((ext_vector_type(8)))  float  v8f;

union AFrag { v16bf v; __bf16 h[16]; };
union CFrag { v8f  v; float  f[8];  };

// A-matrix (16-bit, 16x32) K index for pair j (0..7); kb = (lane>>4)*8
__device__ __forceinline__ int kA(int j, int kb) {
  return (j < 4) ? (kb + 2 * j) : (16 + kb + 2 * (j - 4));
}
// B-matrix (16-bit, 32x16) K index for pair j; hi = lane>>4
__device__ __forceinline__ int kB(int j, int hi) {
  return (j < 4) ? (hi * 8 + 2 * j) : (16 + hi * 8 + 2 * (j - 4));
}

__device__ __forceinline__ v8f wmma_bf16(const v16bf& a, const v16bf& b, const v8f& c) {
  return __builtin_amdgcn_wmma_f32_16x16x32_bf16(false, a, false, b, (short)0, c, false, false);
}

// ---------------- Tensor Data Mover (TDM) support ----------------
#if __has_builtin(__builtin_amdgcn_tensor_load_to_lds) && \
    __has_builtin(__builtin_amdgcn_s_wait_tensorcnt)
#define USE_TDM 1
#else
#define USE_TDM 0
#endif

#if USE_TDM
typedef __attribute__((ext_vector_type(4))) unsigned int u32x4;
typedef __attribute__((ext_vector_type(8))) int          i32x8;
typedef __attribute__((ext_vector_type(4))) int          i32x4;

// 2-D tile load: tile_x (contiguous, elems) x tile_y rows of bf16 from a
// row-major matrix with row stride `stride_elems`, into LDS at lds_addr with
// per-row LDS padding described by pad codes (interval: DWORDs=2<<code,
// amount: DWORDs=code+1).  Descriptor per CDNA5 ISA section 8.3-8.6.
__device__ __forceinline__ void tdm_load_2d_bf16(
    unsigned lds_addr, const void* gptr,
    unsigned tile_x, unsigned tile_y,
    unsigned long long stride_elems,
    unsigned pad_interval_code, unsigned pad_amount_code) {
  unsigned long long ga = (unsigned long long)(size_t)gptr;
  u32x4 g0;
  g0[0] = 1u;                                          // count=1, user desc
  g0[1] = lds_addr;                                    // lds_addr [63:32]
  g0[2] = (unsigned)(ga & 0xFFFFFFFFull);              // global_addr low
  g0[3] = (unsigned)((ga >> 32) & 0x01FFFFFFull)       // global_addr [56:32]
          | (2u << 30);                                // type=2 @ [127:126]
  const unsigned TD = 0x40000000u;                     // huge tensor dims (no OOB)
  i32x8 g1;
  g1[0] = (int)((1u << 16)                             // data_size = 2 bytes
              | (1u << 20)                             // pad_enable
              | (pad_interval_code << 22)
              | (pad_amount_code << 25));
  g1[1] = (int)((TD & 0xFFFFu) << 16);                 // tensor_dim0[15:0]
  g1[2] = (int)((TD >> 16) | ((TD & 0xFFFFu) << 16));  // td0 hi | tensor_dim1 lo
  g1[3] = (int)((TD >> 16) | (tile_x << 16));          // td1 hi | tile_dim0
  g1[4] = (int)tile_y;                                 // tile_dim1 | tile_dim2=0
  g1[5] = (int)(unsigned)(stride_elems & 0xFFFFFFFFull);
  g1[6] = (int)(unsigned)((stride_elems >> 32) & 0xFFFFull);
  g1[7] = 0;
  i32x4 gz = {0, 0, 0, 0};
#if defined(__clang_major__) && (__clang_major__ >= 23)
  i32x8 gz8 = {0, 0, 0, 0, 0, 0, 0, 0};
  __builtin_amdgcn_tensor_load_to_lds(g0, g1, gz, gz, gz8, 0);
#else
  __builtin_amdgcn_tensor_load_to_lds(g0, g1, gz, gz, 0);
#endif
}
#endif  // USE_TDM

// ---------------- conversion / packing ----------------
__global__ void cvt_f32_to_bf16(const float* __restrict__ in, __bf16* __restrict__ out,
                                unsigned long long n) {
  unsigned long long i = (unsigned long long)blockIdx.x * blockDim.x + threadIdx.x;
  if (i < n) out[i] = (__bf16)in[i];
}

// Wq/Wk/Wv are (H, D, DK) f32 -> Wcat (D x 3*D) bf16, columns [Q heads | K heads | V heads]
__global__ void pack_qkv(const float* __restrict__ Wq, const float* __restrict__ Wk,
                         const float* __restrict__ Wv, __bf16* __restrict__ Wcat) {
  unsigned long long i = (unsigned long long)blockIdx.x * blockDim.x + threadIdx.x;
  if (i >= (unsigned long long)Dc * QKVW) return;
  int d = (int)(i / QKVW);
  int j = (int)(i % QKVW);
  int sel = j >> 10;
  int hh  = (j & 1023) >> 6;
  int e   = j & 63;
  const float* W = (sel == 0) ? Wq : ((sel == 1) ? Wk : Wv);
  Wcat[i] = (__bf16)W[((size_t)hh * Dc + d) * DKc + e];
}

// ---------------- tiled bf16 WMMA GEMM ----------------
// C(MxN) = A(MxK,bf16,rowmajor) * B(KxN,bf16,rowmajor), f32 accum.
// EPI 0: store bf16.  EPI 1: +bias, ReLU, row-mask, store bf16.  EPI 2: +bias, row-mask, store f32.
template <int EPI>
__global__ __launch_bounds__(256) void gemm_bf16(
    const __bf16* __restrict__ A, const __bf16* __restrict__ Bm,
    const float* __restrict__ bias, const int* __restrict__ lens,
    __bf16* __restrict__ outb, float* __restrict__ outf,
    int M, int N, int K) {
  __shared__ __align__(16) __bf16 As[2][128][40];    // 32 data + 8 pad per row
  __shared__ __align__(16) __bf16 Bs[2][32][136];    // 128 data + 8 pad per row

  const int tid  = threadIdx.x;
  const int wid  = tid >> 5;
  const int lane = tid & 31;
  const int ml   = lane & 15;
  const int hi   = lane >> 4;
  const int kb   = hi * 8;

  const int m0 = blockIdx.y * 128;
  const int n0 = blockIdx.x * 128;
  const int waveM = (wid >> 2) * 64;   // 2 waves along M
  const int waveN = (wid & 3) * 32;    // 4 waves along N

  CFrag acc[4][2];
#pragma unroll
  for (int rt = 0; rt < 4; ++rt)
#pragma unroll
    for (int ct = 0; ct < 2; ++ct)
#pragma unroll
      for (int r = 0; r < 8; ++r) acc[rt][ct].f[r] = 0.f;

  const int KT = K >> 5;

  auto computeStep = [&](int buf) {
    AFrag af[4], bfr[2];
#pragma unroll
    for (int rt = 0; rt < 4; ++rt) {
      int row = waveM + rt * 16 + ml;
#pragma unroll
      for (int j = 0; j < 8; ++j) {
        int k = kA(j, kb);
        af[rt].h[2 * j]     = As[buf][row][k];
        af[rt].h[2 * j + 1] = As[buf][row][k + 1];
      }
    }
#pragma unroll
    for (int ct = 0; ct < 2; ++ct) {
      int col = waveN + ct * 16 + ml;
#pragma unroll
      for (int j = 0; j < 8; ++j) {
        int k = kB(j, hi);
        bfr[ct].h[2 * j]     = Bs[buf][k][col];
        bfr[ct].h[2 * j + 1] = Bs[buf][k + 1][col];
      }
    }
#pragma unroll
    for (int rt = 0; rt < 4; ++rt)
#pragma unroll
      for (int ct = 0; ct < 2; ++ct)
        acc[rt][ct].v = wmma_bf16(af[rt].v, bfr[ct].v, acc[rt][ct].v);
  };

#if USE_TDM
  // --- Tensor Data Mover pipeline: wave 0 drives the DMA, TENSORcnt + barriers sync ---
  auto issueTiles = [&](int buf, int k0) {
    tdm_load_2d_bf16((unsigned)(size_t)&As[buf][0][0],
                     A + (size_t)m0 * K + k0,
                     /*tile_x=*/32, /*tile_y=*/128, (unsigned long long)K,
                     /*interval 16 DW=*/3, /*amount 4 DW=*/3);
    tdm_load_2d_bf16((unsigned)(size_t)&Bs[buf][0][0],
                     Bm + (size_t)k0 * N + n0,
                     /*tile_x=*/128, /*tile_y=*/32, (unsigned long long)N,
                     /*interval 64 DW=*/5, /*amount 4 DW=*/3);
  };
  if (wid == 0) issueTiles(0, 0);
  for (int kt = 0; kt < KT; ++kt) {
    int buf = kt & 1;
    if (wid == 0) {
      if (kt + 1 < KT) {
        issueTiles(buf ^ 1, (kt + 1) << 5);
        __builtin_amdgcn_s_wait_tensorcnt(2);   // current buf's 2 loads done
      } else {
        __builtin_amdgcn_s_wait_tensorcnt(0);
      }
    }
    __syncthreads();          // publish LDS tiles to all waves
    computeStep(buf);
    __syncthreads();          // all reads of buf done before TDM overwrites it
  }
#else
  // --- fallback: VGPR staging, double buffered ---
  auto loadTiles = [&](int buf, int k0) {
#pragma unroll
    for (int i = 0; i < 2; ++i) {                   // A: 128x32 = 512 chunks
      int ch = tid * 2 + i;
      int row = ch >> 2, col = (ch & 3) * 8;
      *(uint4*)&As[buf][row][col] =
          *(const uint4*)(A + (size_t)(m0 + row) * K + k0 + col);
    }
#pragma unroll
    for (int i = 0; i < 2; ++i) {                   // B: 32x128 = 512 chunks
      int ch = tid * 2 + i;
      int row = ch >> 4, col = (ch & 15) * 8;
      *(uint4*)&Bs[buf][row][col] =
          *(const uint4*)(Bm + (size_t)(k0 + row) * N + n0 + col);
    }
  };
  loadTiles(0, 0);
  __syncthreads();
  for (int kt = 0; kt < KT; ++kt) {
    int buf = kt & 1;
    if (kt + 1 < KT) {
      loadTiles(buf ^ 1, (kt + 1) << 5);
      if (kt + 2 < KT) {
        __builtin_prefetch(A + (size_t)(m0 + (tid >> 1)) * K + ((kt + 2) << 5), 0, 1);
        __builtin_prefetch(Bm + (size_t)(((kt + 2) << 5) + (tid >> 4)) * N + n0, 0, 1);
      }
    }
    computeStep(buf);
    __syncthreads();
  }
#endif

  // epilogue: C layout row = r + 8*hi, col = ml (per 16x16 tile)
#pragma unroll
  for (int rt = 0; rt < 4; ++rt) {
#pragma unroll
    for (int r = 0; r < 8; ++r) {
      int gm = m0 + waveM + rt * 16 + r + 8 * hi;
      bool rowValid = true;
      if (EPI != 0) {
        int s = gm & (Sc - 1);
        int bidx = gm >> 10;
        rowValid = (s < lens[bidx]);
      }
#pragma unroll
      for (int ct = 0; ct < 2; ++ct) {
        int gn = n0 + waveN + ct * 16 + ml;
        float v = acc[rt][ct].f[r];
        if (EPI == 1) {
          v += bias[gn];
          v = v > 0.f ? v : 0.f;
          if (!rowValid) v = 0.f;
          outb[(size_t)gm * N + gn] = (__bf16)v;
        } else if (EPI == 2) {
          v += bias[gn];
          if (!rowValid) v = 0.f;
          outf[(size_t)gm * N + gn] = v;
        } else {
          outb[(size_t)gm * N + gn] = (__bf16)v;
        }
      }
    }
  }
}

// ---------------- flash attention (bf16 WMMA, f32 softmax) ----------------
// QKV: (B, S, 3*D) bf16; Z out: (B, S, D) f32, query rows >= len zeroed.
__global__ __launch_bounds__(256) void attn_kernel(
    const __bf16* __restrict__ QKV, const int* __restrict__ lens,
    float* __restrict__ Z) {
  constexpr int ROW = QKVW;
  __shared__ __align__(16) __bf16 Ks[64][72];        // 64 data + 8 pad per row
  __shared__ __align__(16) __bf16 Vs[64][72];
  __shared__ __align__(16) __bf16 Ps[8][16][72];

  const int tid = threadIdx.x, wid = tid >> 5, lane = tid & 31;
  const int ml = lane & 15, hi = lane >> 4, kb = hi * 8;
  const int b = blockIdx.x >> 4, h = blockIdx.x & 15;
  const int len = lens[b];
  const int q0 = blockIdx.y * 128 + wid * 16;

  const __bf16* Qb = QKV + (size_t)b * Sc * ROW + h * DKc;
  const __bf16* Kb = Qb + Dc;
  const __bf16* Vb = Qb + 2 * Dc;

  // Q A-fragments for this wave's 16 query rows (DK=64 -> 2 K-steps)
  AFrag aq[2];
#pragma unroll
  for (int st = 0; st < 2; ++st)
#pragma unroll
    for (int j = 0; j < 8; ++j) {
      int e = kA(j, kb) + 32 * st;
      const __bf16* p = Qb + (size_t)(q0 + ml) * ROW + e;
      aq[st].h[2 * j]     = p[0];
      aq[st].h[2 * j + 1] = p[1];
    }

  float run_m[8], run_l[8];
  CFrag zacc[4];
#pragma unroll
  for (int r = 0; r < 8; ++r) { run_m[r] = -__builtin_inff(); run_l[r] = 0.f; }
#pragma unroll
  for (int et = 0; et < 4; ++et)
#pragma unroll
    for (int r = 0; r < 8; ++r) zacc[et].f[r] = 0.f;

  for (int kb0 = 0; kb0 < Sc; kb0 += 64) {
    __syncthreads();   // previous iteration's Ks/Vs readers are done
#if USE_TDM
    if (wid == 0) {
      tdm_load_2d_bf16((unsigned)(size_t)&Ks[0][0], Kb + (size_t)kb0 * ROW,
                       /*tile_x=*/64, /*tile_y=*/64, (unsigned long long)ROW,
                       /*interval 32 DW=*/4, /*amount 4 DW=*/3);
      tdm_load_2d_bf16((unsigned)(size_t)&Vs[0][0], Vb + (size_t)kb0 * ROW,
                       /*tile_x=*/64, /*tile_y=*/64, (unsigned long long)ROW,
                       /*interval 32 DW=*/4, /*amount 4 DW=*/3);
      __builtin_amdgcn_s_wait_tensorcnt(0);
    }
#else
#pragma unroll
    for (int i = 0; i < 2; ++i) {           // 64x64 tiles, 512 chunks each
      int ch = tid * 2 + i;
      int row = ch >> 3, col = (ch & 7) * 8;
      *(uint4*)&Ks[row][col] = *(const uint4*)(Kb + (size_t)(kb0 + row) * ROW + col);
      *(uint4*)&Vs[row][col] = *(const uint4*)(Vb + (size_t)(kb0 + row) * ROW + col);
    }
#endif
    __syncthreads();

    // scores: S_tile[nt] = Q(16x64) * K_blk^T, B[e, key] = Ks[key][e]
    CFrag sc[4];
#pragma unroll
    for (int nt = 0; nt < 4; ++nt) {
#pragma unroll
      for (int r = 0; r < 8; ++r) sc[nt].f[r] = 0.f;
#pragma unroll
      for (int st = 0; st < 2; ++st) {
        AFrag bk;
#pragma unroll
        for (int j = 0; j < 8; ++j) {
          int e = kB(j, hi) + 32 * st;
          bk.h[2 * j]     = Ks[nt * 16 + ml][e];
          bk.h[2 * j + 1] = Ks[nt * 16 + ml][e + 1];
        }
        sc[nt].v = wmma_bf16(aq[st].v, bk.v, sc[nt].v);
      }
    }
    // key mask + 1/sqrt(dk) scale
#pragma unroll
    for (int nt = 0; nt < 4; ++nt) {
      bool kvld = (kb0 + nt * 16 + ml) < len;
#pragma unroll
      for (int r = 0; r < 8; ++r)
        sc[nt].f[r] = kvld ? sc[nt].f[r] * 0.125f : -__builtin_inff();
    }
    // online softmax per query row (reduce across 16-lane half)
#pragma unroll
    for (int r = 0; r < 8; ++r) {
      float mx = fmaxf(fmaxf(sc[0].f[r], sc[1].f[r]), fmaxf(sc[2].f[r], sc[3].f[r]));
#pragma unroll
      for (int off = 1; off < 16; off <<= 1) mx = fmaxf(mx, __shfl_xor(mx, off, 32));
      float nm = fmaxf(run_m[r], mx);
      float scale = __expf(run_m[r] - nm);
      float ps = 0.f;
#pragma unroll
      for (int nt = 0; nt < 4; ++nt) {
        float p = __expf(sc[nt].f[r] - nm);
        sc[nt].f[r] = p;
        ps += p;
      }
#pragma unroll
      for (int off = 1; off < 16; off <<= 1) ps += __shfl_xor(ps, off, 32);
      run_l[r] = run_l[r] * scale + ps;
      run_m[r] = nm;
#pragma unroll
      for (int et = 0; et < 4; ++et) zacc[et].f[r] *= scale;
    }
    // transpose P through LDS into A-fragment layout
#pragma unroll
    for (int nt = 0; nt < 4; ++nt)
#pragma unroll
      for (int r = 0; r < 8; ++r)
        Ps[wid][r + 8 * hi][nt * 16 + ml] = (__bf16)sc[nt].f[r];
    __syncthreads();

    AFrag ap[2];
#pragma unroll
    for (int st = 0; st < 2; ++st)
#pragma unroll
      for (int j = 0; j < 8; ++j) {
        int k = kA(j, kb) + 32 * st;
        ap[st].h[2 * j]     = Ps[wid][ml][k];
        ap[st].h[2 * j + 1] = Ps[wid][ml][k + 1];
      }
    // z += P(16x64) * V(64x64), B[key, e] = Vs[key][e]
#pragma unroll
    for (int et = 0; et < 4; ++et) {
#pragma unroll
      for (int st = 0; st < 2; ++st) {
        AFrag bv;
#pragma unroll
        for (int j = 0; j < 8; ++j) {
          int k = kB(j, hi) + 32 * st;
          bv.h[2 * j]     = Vs[k][et * 16 + ml];
          bv.h[2 * j + 1] = Vs[k + 1][et * 16 + ml];
        }
        zacc[et].v = wmma_bf16(ap[st].v, bv.v, zacc[et].v);
      }
    }
  }

#pragma unroll
  for (int r = 0; r < 8; ++r) {
    int q = q0 + r + 8 * hi;
    float inv = (q < len) ? 1.f / run_l[r] : 0.f;   // query mask
#pragma unroll
    for (int et = 0; et < 4; ++et)
      Z[((size_t)b * Sc + q) * Dc + h * DKc + et * 16 + ml] = zacc[et].f[r] * inv;
  }
}

// ---------------- residual + LayerNorm (one row per block) ----------------
__global__ __launch_bounds__(256) void add_ln_kernel(
    const float* __restrict__ X, const float* __restrict__ R,
    const float* __restrict__ w, const float* __restrict__ bb,
    float* __restrict__ outf, __bf16* __restrict__ outb) {
  __shared__ float red0[8], red1[8];
  const int row = blockIdx.x, tid = threadIdx.x, wid = tid >> 5, lane = tid & 31;
  const float* xr = X + (size_t)row * Dc;
  const float* rr = R + (size_t)row * Dc;
  float v[4], s = 0.f, s2 = 0.f;
#pragma unroll
  for (int i = 0; i < 4; ++i) {
    int c = tid + 256 * i;
    float t = xr[c] + rr[c];
    v[i] = t; s += t; s2 += t * t;
  }
#pragma unroll
  for (int off = 16; off > 0; off >>= 1) {
    s  += __shfl_xor(s, off, 32);
    s2 += __shfl_xor(s2, off, 32);
  }
  if (lane == 0) { red0[wid] = s; red1[wid] = s2; }
  __syncthreads();
  float ts = 0.f, ts2 = 0.f;
#pragma unroll
  for (int i = 0; i < 8; ++i) { ts += red0[i]; ts2 += red1[i]; }
  float mu  = ts / Dc;
  float var = ts2 / Dc - mu * mu;
  float rs  = rsqrtf(var + 1e-5f);
#pragma unroll
  for (int i = 0; i < 4; ++i) {
    int c = tid + 256 * i;
    float y = (v[i] - mu) * rs * w[c] + bb[c];
    outf[(size_t)row * Dc + c] = y;
    if (outb) outb[(size_t)row * Dc + c] = (__bf16)y;
  }
}

// ---------------- host launcher ----------------
extern "C" void kernel_launch(void* const* d_in, const int* in_sizes, int n_in,
                              void* d_out, int out_size, void* d_ws, size_t ws_size,
                              hipStream_t stream) {
  const float* x     = (const float*)d_in[0];
  const float* Wq    = (const float*)d_in[1];
  const float* Wk    = (const float*)d_in[2];
  const float* Wv    = (const float*)d_in[3];
  const float* fc1_w = (const float*)d_in[4];
  const float* fc1_b = (const float*)d_in[5];
  const float* fc2_w = (const float*)d_in[6];
  const float* fc2_b = (const float*)d_in[7];
  const float* ln1_w = (const float*)d_in[8];
  const float* ln1_b = (const float*)d_in[9];
  const float* ln2_w = (const float*)d_in[10];
  const float* ln2_b = (const float*)d_in[11];
  const int*   lens  = (const int*)d_in[12];
  float* out = (float*)d_out;

  char* ws = (char*)d_ws;
  size_t off = 0;
  auto alloc = [&](size_t bytes) -> void* {
    void* p = ws + off;
    off += (bytes + 255) & ~(size_t)255;
    return p;
  };
  __bf16* Xb   = (__bf16*)alloc((size_t)Mrows * Dc * 2);       // x in bf16
  __bf16* Wcat = (__bf16*)alloc((size_t)Dc * QKVW * 2);        // fused QKV weight
  __bf16* W1b  = (__bf16*)alloc((size_t)Dc * Dc * 2);
  __bf16* W2b  = (__bf16*)alloc((size_t)Dc * Dc * 2);
  __bf16* QKV  = (__bf16*)alloc((size_t)Mrows * QKVW * 2);
  float*  Zf   = (float*)alloc((size_t)Mrows * Dc * 4);
  float*  X1f  = (float*)alloc((size_t)Mrows * Dc * 4);
  __bf16* X1b  = (__bf16*)alloc((size_t)Mrows * Dc * 2);
  __bf16* FFb  = (__bf16*)alloc((size_t)Mrows * Dc * 2);
  float*  FF2f = (float*)alloc((size_t)Mrows * Dc * 4);
  (void)ws_size; (void)in_sizes; (void)n_in; (void)out_size;

  // 1) conversions / packing
  {
    unsigned long long n = (unsigned long long)Mrows * Dc;
    cvt_f32_to_bf16<<<dim3((unsigned)((n + 255) / 256)), 256, 0, stream>>>(x, Xb, n);
  }
  {
    unsigned long long n = (unsigned long long)Dc * QKVW;
    pack_qkv<<<dim3((unsigned)((n + 255) / 256)), 256, 0, stream>>>(Wq, Wk, Wv, Wcat);
  }
  {
    unsigned long long n = (unsigned long long)Dc * Dc;
    cvt_f32_to_bf16<<<dim3((unsigned)((n + 255) / 256)), 256, 0, stream>>>(fc1_w, W1b, n);
    cvt_f32_to_bf16<<<dim3((unsigned)((n + 255) / 256)), 256, 0, stream>>>(fc2_w, W2b, n);
  }

  // 2) fused QKV projection: (8192 x 1024) @ (1024 x 3072) -> bf16
  gemm_bf16<0><<<dim3(QKVW / 128, Mrows / 128), 256, 0, stream>>>(
      Xb, Wcat, nullptr, nullptr, QKV, nullptr, Mrows, QKVW, Dc);

  // 3) attention -> Z (f32, query-masked)
  attn_kernel<<<dim3(Bc * Hc, Sc / 128), 256, 0, stream>>>(QKV, lens, Zf);

  // 4) x1 = LN(x + z)  (f32 + bf16 copy)
  add_ln_kernel<<<Mrows, 256, 0, stream>>>(x, Zf, ln1_w, ln1_b, X1f, X1b);

  // 5) ff = relu(x1 @ fc1 + b1) * mask  -> bf16
  gemm_bf16<1><<<dim3(Dc / 128, Mrows / 128), 256, 0, stream>>>(
      X1b, W1b, fc1_b, lens, FFb, nullptr, Mrows, Dc, Dc);

  // 6) ff2 = (ff @ fc2 + b2) * mask -> f32
  gemm_bf16<2><<<dim3(Dc / 128, Mrows / 128), 256, 0, stream>>>(
      FFb, W2b, fc2_b, lens, nullptr, FF2f, Mrows, Dc, Dc);

  // 7) out = LN(x1 + ff2)
  add_ln_kernel<<<Mrows, 256, 0, stream>>>(X1f, FF2f, ln2_w, ln2_b, out, nullptr);
}
